// QLSTM_65481071408629
// MI455X (gfx1250) — matrix operand order repaired and checked by reading, not compiled, not run on hardware
//
#include <hip/hip_runtime.h>
#include <hip/hip_bf16.h>
#include <math.h>

typedef __attribute__((ext_vector_type(16))) __bf16 v16bf;
typedef __attribute__((ext_vector_type(8)))  __bf16 v8bf;
typedef __attribute__((ext_vector_type(8)))  float  v8f;
typedef __attribute__((ext_vector_type(4)))  float  v4f;

#define Tn 256
#define Bn 128
#define Dn 1024
#define Hn 1024
#define DHn 2048                   // D + H
#define BH (Bn * Hn)               // 131072
#define NTILES (Hn / 16)           // 64 recurrent blocks
#define WELEMS (4ull * Hn * DHn)   // bf16 weight elements
#define KC (Hn / 32)               // 32 K-chunks in recurrent GEMM

__device__ __forceinline__ float sigmoidf_(float x) {
  return 1.0f / (1.0f + expf(-x));
}

// ---------------- weight fp32 -> bf16 conversion ----------------
__global__ __launch_bounds__(256) void cvt_weights(
    const float* __restrict__ Wf, const float* __restrict__ Wi,
    const float* __restrict__ Wg, const float* __restrict__ Wo,
    __bf16* __restrict__ out) {
  const size_t per = (size_t)Hn * DHn;
  size_t base = ((size_t)blockIdx.x * blockDim.x + threadIdx.x) * 4;
  size_t gate = base / per;
  size_t off  = base % per;
  const float* W = (gate == 0) ? Wf : (gate == 1) ? Wi : (gate == 2) ? Wg : Wo;
  v4f v = *(const v4f*)(W + off);
  __bf16* o = out + base;
  o[0] = (__bf16)v.x; o[1] = (__bf16)v.y; o[2] = (__bf16)v.z; o[3] = (__bf16)v.w;
}

// ---------------- state init (path A: persistent) ----------------
__global__ __launch_bounds__(256) void init_stateA(__bf16* __restrict__ hx,
                                                   unsigned* __restrict__ counter) {
  int i = blockIdx.x * blockDim.x + threadIdx.x;
  if (i < BH) hx[i] = (__bf16)0.0f;
  if (i == 0) *counter = 0u;
}

// ---------------- state init (path C: fallback) ----------------
__global__ __launch_bounds__(256) void init_state(__bf16* __restrict__ hx,
                                                  float* __restrict__ cx) {
  int i = blockIdx.x * blockDim.x + threadIdx.x;
  if (i < BH) { hx[i] = (__bf16)0.0f; cx[i] = 0.0f; }
}

// =======================================================================
// PATH A, phase 1: xpre[t][g][b][h] = x[t,b,:] @ Wg[:, 0:D]^T + bias_g
// Fully parallel throughput GEMM.
// =======================================================================
__global__ __launch_bounds__(256) void xproj(
    const float*  __restrict__ x,    // [T*B, D] fp32
    const __bf16* __restrict__ Wb,   // [4][H][DH] bf16 (x-part = cols 0..D-1)
    const float*  __restrict__ bfp, const float* __restrict__ bip,
    const float*  __restrict__ bgp, const float* __restrict__ bop,
    float* __restrict__ xpre) {      // [T][4][B][H] fp32
  const int lane = threadIdx.x & 31;
  const int wid  = threadIdx.x >> 5;
  const int l16  = lane & 15;
  const bool hi  = lane >= 16;

  const int rowtile = blockIdx.x >> 3;                 // 0..2047 (T*B/16)
  const int ntile   = ((blockIdx.x & 7) << 3) + wid;   // 0..63
  const int mrow = (rowtile << 4) + l16;
  const int nrow = (ntile << 4) + l16;

  const size_t per = (size_t)Hn * DHn;
  const __bf16* w0 = Wb + 0 * per + (size_t)nrow * DHn;
  const __bf16* w1 = Wb + 1 * per + (size_t)nrow * DHn;
  const __bf16* w2 = Wb + 2 * per + (size_t)nrow * DHn;
  const __bf16* w3 = Wb + 3 * per + (size_t)nrow * DHn;

  v8f acc0 = {}, acc1 = {}, acc2 = {}, acc3 = {};

  const float* xrow = x + (size_t)mrow * Dn;
  for (int k0 = 0; k0 < Dn; k0 += 32) {
    const int ka = k0 + (hi ? 16 : 0);
    v4f f0 = *(const v4f*)(xrow + ka);
    v4f f1 = *(const v4f*)(xrow + ka + 4);
    v4f f2 = *(const v4f*)(xrow + ka + 8);
    v4f f3 = *(const v4f*)(xrow + ka + 12);
    v16bf a;
    a[0]  = (__bf16)f0.x; a[1]  = (__bf16)f0.y; a[2]  = (__bf16)f0.z; a[3]  = (__bf16)f0.w;
    a[4]  = (__bf16)f1.x; a[5]  = (__bf16)f1.y; a[6]  = (__bf16)f1.z; a[7]  = (__bf16)f1.w;
    a[8]  = (__bf16)f2.x; a[9]  = (__bf16)f2.y; a[10] = (__bf16)f2.z; a[11] = (__bf16)f2.w;
    a[12] = (__bf16)f3.x; a[13] = (__bf16)f3.y; a[14] = (__bf16)f3.z; a[15] = (__bf16)f3.w;

    const int kb = k0 + (hi ? 16 : 0);
    v16bf b0 = *(const v16bf*)(w0 + kb);
    v16bf b1 = *(const v16bf*)(w1 + kb);
    v16bf b2 = *(const v16bf*)(w2 + kb);
    v16bf b3 = *(const v16bf*)(w3 + kb);
    acc0 = __builtin_amdgcn_wmma_f32_16x16x32_bf16(false, a, false, b0, (short)0, acc0, false, false);
    acc1 = __builtin_amdgcn_wmma_f32_16x16x32_bf16(false, a, false, b1, (short)0, acc1, false, false);
    acc2 = __builtin_amdgcn_wmma_f32_16x16x32_bf16(false, a, false, b2, (short)0, acc2, false, false);
    acc3 = __builtin_amdgcn_wmma_f32_16x16x32_bf16(false, a, false, b3, (short)0, acc3, false, false);
  }

  const int col = nrow;
  const float bfv = bfp[col], biv = bip[col], bgv = bgp[col], bov = bop[col];
  const int t     = rowtile >> 3;
  const int btile = rowtile & 7;
  const int rbase = (btile << 4) + (hi ? 8 : 0);
  float* xp = xpre + (size_t)t * 4 * BH;
#pragma unroll
  for (int v = 0; v < 8; ++v) {
    const size_t idx = (size_t)(rbase + v) * Hn + col;
    xp[0 * BH + idx] = acc0[v] + bfv;
    xp[1 * BH + idx] = acc1[v] + biv;
    xp[2 * BH + idx] = acc2[v] + bgv;
    xp[3 * BH + idx] = acc3[v] + bov;
  }
}

// =======================================================================
// PATH A, phase 2: persistent recurrent kernel. 64 blocks x 8 waves.
// Loop-invariant recurrent weights staged once into 128KB of LDS,
// pre-swizzled into per-lane WMMA B-operand chunks; cell state in regs;
// hx double-buffered in ws; one agent-scope grid barrier per step.
// =======================================================================
__global__ __launch_bounds__(256) void lstm_recur(
    const float*  __restrict__ xpre,   // [T][4][B][H]
    const __bf16* __restrict__ Wb,     // [4][H][DH]
    __bf16* __restrict__ hxA, __bf16* __restrict__ hxB,
    unsigned* __restrict__ counter,
    float* __restrict__ out) {         // [T*B*H + 2*B*H]
  // LDS: [gate][kc][lane] 16-half chunks: 4*32*32*16 bf16 = 128KB
  __shared__ __bf16 ldsW[4 * KC * 32 * 16];

  const int lane  = threadIdx.x & 31;
  const int mtile = threadIdx.x >> 5;  // 0..7
  const int ntile = blockIdx.x;        // 0..63
  const int l16   = lane & 15;
  const bool hi   = lane >= 16;

  const int mrow = (mtile << 4) + l16;
  const int nrow = (ntile << 4) + l16;
  const size_t per = (size_t)Hn * DHn;

  // ---- stage recurrent weight halves (cols D..D+H-1) into LDS ----
  for (int c = threadIdx.x; c < 4 * KC * 32; c += 256) {
    const int g   = c >> 10;          // /(KC*32)
    const int rem = c & 1023;
    const int kc  = rem >> 5;
    const int ln  = rem & 31;
    const int row = (ntile << 4) + (ln & 15);
    const int kb  = Dn + (kc << 5) + ((ln & 16) ? 16 : 0);
    v16bf v = *(const v16bf*)(Wb + (size_t)g * per + (size_t)row * DHn + kb);
    *(v16bf*)&ldsW[(size_t)c * 16] = v;
  }
  __syncthreads();

  const __bf16* lw = &ldsW[lane * 16];  // per-lane chunk base
  const int col   = nrow;
  const int rbase = (mtile << 4) + (hi ? 8 : 0);

  float c[8];
#pragma unroll
  for (int v = 0; v < 8; ++v) c[v] = 0.0f;

  for (int t = 0; t < Tn; ++t) {
    const __bf16* hin  = (t & 1) ? hxB : hxA;
    __bf16*       hout = (t & 1) ? hxA : hxB;

    // hoist xpre gate loads: independent of the GEMM, hide under K-loop
    const float* xp = xpre + (size_t)t * 4 * BH;
    float xf[8], xi[8], xg[8], xo[8];
#pragma unroll
    for (int v = 0; v < 8; ++v) {
      const size_t idx = (size_t)(rbase + v) * Hn + col;
      xf[v] = xp[0 * BH + idx];
      xi[v] = xp[1 * BH + idx];
      xg[v] = xp[2 * BH + idx];
      xo[v] = xp[3 * BH + idx];
    }

    v8f acc0 = {}, acc1 = {}, acc2 = {}, acc3 = {};
    const __bf16* hrow = hin + (size_t)mrow * Hn;
    for (int kc = 0; kc < KC; ++kc) {
      const int ka = (kc << 5) + (hi ? 16 : 0);
      union { v16bf v; v8bf h[2]; } a;
      a.h[0] = *(const v8bf*)(hrow + ka);
      a.h[1] = *(const v8bf*)(hrow + ka + 8);

      // B operands from LDS: offset (g*KC + kc)*512 halfs from lane base
      v16bf b0 = *(const v16bf*)(lw + (size_t)(0 * KC + kc) * 512);
      v16bf b1 = *(const v16bf*)(lw + (size_t)(1 * KC + kc) * 512);
      v16bf b2 = *(const v16bf*)(lw + (size_t)(2 * KC + kc) * 512);
      v16bf b3 = *(const v16bf*)(lw + (size_t)(3 * KC + kc) * 512);
      acc0 = __builtin_amdgcn_wmma_f32_16x16x32_bf16(false, a.v, false, b0, (short)0, acc0, false, false);
      acc1 = __builtin_amdgcn_wmma_f32_16x16x32_bf16(false, a.v, false, b1, (short)0, acc1, false, false);
      acc2 = __builtin_amdgcn_wmma_f32_16x16x32_bf16(false, a.v, false, b2, (short)0, acc2, false, false);
      acc3 = __builtin_amdgcn_wmma_f32_16x16x32_bf16(false, a.v, false, b3, (short)0, acc3, false, false);
    }

    float* out_t = out + (size_t)t * BH;
#pragma unroll
    for (int v = 0; v < 8; ++v) {
      const size_t idx = (size_t)(rbase + v) * Hn + col;
      const float fg = sigmoidf_(acc0[v] + xf[v]);
      const float ig = sigmoidf_(acc1[v] + xi[v]);
      const float gg = tanhf    (acc2[v] + xg[v]);
      const float og = sigmoidf_(acc3[v] + xo[v]);
      const float cn = fg * c[v] + ig * gg;
      c[v] = cn;
      const float h = og * tanhf(cn);
      out_t[idx]  = h;
      hout[idx]   = (__bf16)h;
      if (t == Tn - 1) {                  // (hx, cx) tail of d_out
        out[(size_t)Tn * BH + idx]      = h;
        out[(size_t)Tn * BH + BH + idx] = cn;
      }
    }

    // ---- grid barrier: release, arrive, spin, acquire ----
    __builtin_amdgcn_fence(__ATOMIC_RELEASE, "agent");
    __syncthreads();
    if (threadIdx.x == 0) {
      __hip_atomic_fetch_add(counter, 1u, __ATOMIC_RELAXED, __HIP_MEMORY_SCOPE_AGENT);
      const unsigned tgt = (unsigned)(t + 1) * (unsigned)NTILES;
      while (__hip_atomic_load(counter, __ATOMIC_RELAXED, __HIP_MEMORY_SCOPE_AGENT) < tgt)
        __builtin_amdgcn_s_sleep(1);
    }
    __syncthreads();
    __builtin_amdgcn_fence(__ATOMIC_ACQUIRE, "agent");
  }
}

// =======================================================================
// PATH C fallback (small ws): per-step launches, fused K=2048.
// =======================================================================
__global__ __launch_bounds__(256) void lstm_step(
    const float*  __restrict__ x_t,
    const __bf16* __restrict__ Wb,
    const float*  __restrict__ bfp, const float* __restrict__ bip,
    const float*  __restrict__ bgp, const float* __restrict__ bop,
    const __bf16* __restrict__ hx_in,
    __bf16*       __restrict__ hx_out,
    float*        __restrict__ cx,
    float*        __restrict__ out_t) {
  const int lane  = threadIdx.x & 31;
  const int mtile = threadIdx.x >> 5;
  const int ntile = blockIdx.x;
  const int l16   = lane & 15;
  const bool hi   = lane >= 16;

  const int mrow = (mtile << 4) + l16;
  const int nrow = (ntile << 4) + l16;

  const size_t per = (size_t)Hn * DHn;
  const __bf16* w0 = Wb + 0 * per + (size_t)nrow * DHn;
  const __bf16* w1 = Wb + 1 * per + (size_t)nrow * DHn;
  const __bf16* w2 = Wb + 2 * per + (size_t)nrow * DHn;
  const __bf16* w3 = Wb + 3 * per + (size_t)nrow * DHn;

  v8f acc0 = {}, acc1 = {}, acc2 = {}, acc3 = {};

  const float* xrow = x_t + (size_t)mrow * Dn;
  for (int k0 = 0; k0 < Dn; k0 += 32) {
    const int ka = k0 + (hi ? 16 : 0);
    v4f f0 = *(const v4f*)(xrow + ka);
    v4f f1 = *(const v4f*)(xrow + ka + 4);
    v4f f2 = *(const v4f*)(xrow + ka + 8);
    v4f f3 = *(const v4f*)(xrow + ka + 12);
    v16bf a;
    a[0]  = (__bf16)f0.x; a[1]  = (__bf16)f0.y; a[2]  = (__bf16)f0.z; a[3]  = (__bf16)f0.w;
    a[4]  = (__bf16)f1.x; a[5]  = (__bf16)f1.y; a[6]  = (__bf16)f1.z; a[7]  = (__bf16)f1.w;
    a[8]  = (__bf16)f2.x; a[9]  = (__bf16)f2.y; a[10] = (__bf16)f2.z; a[11] = (__bf16)f2.w;
    a[12] = (__bf16)f3.x; a[13] = (__bf16)f3.y; a[14] = (__bf16)f3.z; a[15] = (__bf16)f3.w;

    const int kb = k0 + (hi ? 16 : 0);
    v16bf b0 = *(const v16bf*)(w0 + kb);
    v16bf b1 = *(const v16bf*)(w1 + kb);
    v16bf b2 = *(const v16bf*)(w2 + kb);
    v16bf b3 = *(const v16bf*)(w3 + kb);
    acc0 = __builtin_amdgcn_wmma_f32_16x16x32_bf16(false, a, false, b0, (short)0, acc0, false, false);
    acc1 = __builtin_amdgcn_wmma_f32_16x16x32_bf16(false, a, false, b1, (short)0, acc1, false, false);
    acc2 = __builtin_amdgcn_wmma_f32_16x16x32_bf16(false, a, false, b2, (short)0, acc2, false, false);
    acc3 = __builtin_amdgcn_wmma_f32_16x16x32_bf16(false, a, false, b3, (short)0, acc3, false, false);
  }

  const __bf16* hrow = hx_in + (size_t)mrow * Hn;
  for (int k0 = 0; k0 < Hn; k0 += 32) {
    const int ka = k0 + (hi ? 16 : 0);
    union { v16bf v; v8bf h[2]; } a;
    a.h[0] = *(const v8bf*)(hrow + ka);
    a.h[1] = *(const v8bf*)(hrow + ka + 8);

    const int kb = Dn + k0 + (hi ? 16 : 0);
    v16bf b0 = *(const v16bf*)(w0 + kb);
    v16bf b1 = *(const v16bf*)(w1 + kb);
    v16bf b2 = *(const v16bf*)(w2 + kb);
    v16bf b3 = *(const v16bf*)(w3 + kb);
    acc0 = __builtin_amdgcn_wmma_f32_16x16x32_bf16(false, a.v, false, b0, (short)0, acc0, false, false);
    acc1 = __builtin_amdgcn_wmma_f32_16x16x32_bf16(false, a.v, false, b1, (short)0, acc1, false, false);
    acc2 = __builtin_amdgcn_wmma_f32_16x16x32_bf16(false, a.v, false, b2, (short)0, acc2, false, false);
    acc3 = __builtin_amdgcn_wmma_f32_16x16x32_bf16(false, a.v, false, b3, (short)0, acc3, false, false);
  }

  const int col = nrow;
  const float bfv = bfp[col], biv = bip[col], bgv = bgp[col], bov = bop[col];
  const int rbase = (mtile << 4) + (hi ? 8 : 0);
#pragma unroll
  for (int v = 0; v < 8; ++v) {
    const size_t idx = (size_t)(rbase + v) * Hn + col;
    const float fg = sigmoidf_(acc0[v] + bfv);
    const float ig = sigmoidf_(acc1[v] + biv);
    const float gg = tanhf(acc2[v] + bgv);
    const float og = sigmoidf_(acc3[v] + bov);
    const float cn = fg * cx[idx] + ig * gg;
    cx[idx] = cn;
    const float h = og * tanhf(cn);
    out_t[idx]  = h;
    hx_out[idx] = (__bf16)h;
  }
}

__global__ __launch_bounds__(256) void finalize(const float* __restrict__ last_h,
                                                const float* __restrict__ cx,
                                                float* __restrict__ out_tail) {
  int i = blockIdx.x * blockDim.x + threadIdx.x;
  if (i < BH) {
    out_tail[i]      = last_h[i];
    out_tail[BH + i] = cx[i];
  }
}

extern "C" void kernel_launch(void* const* d_in, const int* in_sizes, int n_in,
                              void* d_out, int out_size, void* d_ws, size_t ws_size,
                              hipStream_t stream) {
  const float* inputs = (const float*)d_in[0];
  const float* Wf  = (const float*)d_in[1];
  const float* bf_ = (const float*)d_in[2];
  const float* Wi  = (const float*)d_in[3];
  const float* bi_ = (const float*)d_in[4];
  const float* Wg  = (const float*)d_in[5];
  const float* bg_ = (const float*)d_in[6];
  const float* Wo  = (const float*)d_in[7];
  const float* bo_ = (const float*)d_in[8];
  float* out = (float*)d_out;

  // ---- ws layout (path A) ----
  char* base = (char*)d_ws;
  size_t off = 0;
  __bf16* Wb  = (__bf16*)(base + off); off += WELEMS * 2;
  __bf16* hxA = (__bf16*)(base + off); off += (size_t)BH * 2;
  __bf16* hxB = (__bf16*)(base + off); off += (size_t)BH * 2;
  off = (off + 255) & ~(size_t)255;
  unsigned* counter = (unsigned*)(base + off); off += 256;
  float* xpre = (float*)(base + off); off += (size_t)Tn * 4 * BH * 4;
  const size_t needA = off;

  cvt_weights<<<8192, 256, 0, stream>>>(Wf, Wi, Wg, Wo, Wb);

  if (ws_size >= needA) {
    // ===== Path A: hoisted x-projection + persistent recurrence =====
    init_stateA<<<(BH + 255) / 256, 256, 0, stream>>>(hxA, counter);
    xproj<<<(Tn * Bn / 16) * 8, 256, 0, stream>>>(
        inputs, Wb, bf_, bi_, bg_, bo_, xpre);
    lstm_recur<<<NTILES, 256, 0, stream>>>(xpre, Wb, hxA, hxB, counter, out);
  } else {
    // ===== Path C: per-step launches, fused K=2048 =====
    float* cx = (float*)(base + (size_t)(WELEMS * 2) + 2 * (size_t)BH * 2);
    init_state<<<(BH + 255) / 256, 256, 0, stream>>>(hxA, cx);
    for (int t = 0; t < Tn; ++t) {
      const __bf16* hin  = (t & 1) ? hxB : hxA;
      __bf16*       hout = (t & 1) ? hxA : hxB;
      lstm_step<<<NTILES, 256, 0, stream>>>(
          inputs + (size_t)t * Bn * Dn, Wb, bf_, bi_, bg_, bo_,
          hin, hout, cx, out + (size_t)t * BH);
    }
    finalize<<<(BH + 255) / 256, 256, 0, stream>>>(
        out + (size_t)(Tn - 1) * BH, cx, out + (size_t)Tn * BH);
  }
}
